// TriangleCrossAttention_53412213293853
// MI455X (gfx1250) — compile-verified, matrix operand order checked by reading, not run on hardware
//
#include <hip/hip_runtime.h>
#include <hip/hip_bf16.h>
#include <hip/hip_fp16.h>

// ---------------------------------------------------------------------------
// TriangleCrossAttention on gfx1250 (CDNA5)
//  - all GEMMs on V_WMMA_F32_16X16X32_F16 (f32 accumulate)
//  - weights staged to LDS with GLOBAL_LOAD_ASYNC_TO_LDS_B128 (ASYNCcnt path)
//  - 16x64 C strip per wave; fused q/g GEMM reads xn once, 8 WMMAs per A frag
// B=1, N=M=256, CQ=CKV=128, H=4, CH=32, CB=16
// ---------------------------------------------------------------------------

typedef _Float16 h16;
typedef __attribute__((ext_vector_type(16))) _Float16 v16h;
typedef __attribute__((ext_vector_type(8)))  _Float16 v8h;
typedef __attribute__((ext_vector_type(4)))  _Float16 v4h;
typedef __attribute__((ext_vector_type(8)))  float    v8f;

#define WMMA_F16(a, b, c) \
  __builtin_amdgcn_wmma_f32_16x16x32_f16(false, (a), false, (b), (short)0, (c), false, false)

static __device__ __forceinline__ int laneId() { return threadIdx.x & 31; }

// Async copy 16B global -> LDS (CDNA5 GLOBAL_LOAD_ASYNC_TO_LDS_B128, ASYNCcnt).
static __device__ __forceinline__ void async_load_b128(unsigned int lds_off,
                                                       unsigned long long gaddr) {
  asm volatile("global_load_async_to_lds_b128 %0, %1, off"
               :: "v"(lds_off), "v"(gaddr)
               : "memory");
}
static __device__ __forceinline__ void wait_async0() {
  asm volatile("s_wait_asynccnt 0x0" ::: "memory");
}

// Load one 16x32 f16 fragment (A-matrix layout; also valid for the B operand
// when B is a row-major weight matrix W with K contiguous, since
// B[k][n] = W[n][k] mirrors the A striping with M<->N swapped).
// CDNA5 layout: lanes 0-15 -> row M=lane, K 0..7 (v0-3) and 16..23 (v4-7);
// lanes 16-31 -> row M=lane-16, K 8..15 and 24..31.
static __device__ __forceinline__ v16h load_frag(const h16* __restrict__ base,
                                                 int rowTile, int ld, int kbase) {
  const int lane = laneId();
  const h16* p = base + (size_t)(rowTile + (lane & 15)) * ld
                      + kbase + ((lane >> 4) << 3);
  v8h lo = *(const v8h*)(p);
  v8h hi = *(const v8h*)(p + 16);
  v16h f;
#pragma unroll
  for (int i = 0; i < 8; ++i) { f[i] = lo[i]; f[i + 8] = hi[i]; }
  return f;
}

// ---------------------------------------------------------------------------
// 1) LayerNorm over channel dim (128), output f16. One wave per row.
// ---------------------------------------------------------------------------
__global__ void layernorm_kernel(const float* __restrict__ x,
                                 const float* __restrict__ ln_w,
                                 const float* __restrict__ ln_b,
                                 h16* __restrict__ xn) {
  const int wave = threadIdx.x >> 5;
  const int lane = laneId();
  const size_t row = (size_t)blockIdx.x * 8 + wave;   // 0..65535
  const int c = lane << 2;                             // 4 channels per lane

  float4 xv = *(const float4*)(x + row * 128 + c);
  float s  = xv.x + xv.y + xv.z + xv.w;
  float ss = xv.x * xv.x + xv.y * xv.y + xv.z * xv.z + xv.w * xv.w;
#pragma unroll
  for (int off = 16; off >= 1; off >>= 1) {
    s  += __shfl_xor(s,  off);
    ss += __shfl_xor(ss, off);
  }
  const float mu   = s * (1.0f / 128.0f);
  const float var  = ss * (1.0f / 128.0f) - mu * mu;
  const float rstd = rsqrtf(var + 1e-5f);

  float4 wv = *(const float4*)(ln_w + c);
  float4 bv = *(const float4*)(ln_b + c);
  v4h o;
  o[0] = (h16)((xv.x - mu) * rstd * wv.x + bv.x);
  o[1] = (h16)((xv.y - mu) * rstd * wv.y + bv.y);
  o[2] = (h16)((xv.z - mu) * rstd * wv.z + bv.z);
  o[3] = (h16)((xv.w - mu) * rstd * wv.w + bv.w);
  *(v4h*)(xn + row * 128 + c) = o;
}

// ---------------------------------------------------------------------------
// 2) f32 -> f16 conversion (weights, z)
// ---------------------------------------------------------------------------
__global__ void cvt_f16_kernel(const float* __restrict__ in,
                               h16* __restrict__ out, int n) {
  int i = blockIdx.x * blockDim.x + threadIdx.x;
  if (i < n) out[i] = (h16)in[i];
}

// ---------------------------------------------------------------------------
// 3) Column mean over inner N (q) of xn -> xbar (256x128, f16)
// ---------------------------------------------------------------------------
__global__ void colmean_kernel(const h16* __restrict__ xn,
                               h16* __restrict__ xbar) {
  const int n = blockIdx.x;        // 0..255
  const int c = threadIdx.x;       // 0..127
  float acc = 0.0f;
  for (int q = 0; q < 256; ++q)
    acc += (float)xn[((size_t)(n * 256 + q)) * 128 + c];
  xbar[n * 128 + c] = (h16)(acc * (1.0f / 256.0f));
}

// ---------------------------------------------------------------------------
// 4a) WMMA GEMM: C[M,N] = act(alpha * A[M,K] * W[N,K]^T + bias)
//     W (<=128x128 f16, ldw==K) async-staged to LDS once per block.
//     Each wave computes a 16x64 strip; all 4 B frags batched before the
//     WMMA chain so DS latency is paid once per k-step.
//     Requires: K % 32 == 0, N % 64 == 0, M % 16 == 0.
// ---------------------------------------------------------------------------
#define GF_SIGMOID   1
#define GF_TRANSPOSE 2   // store C^T (Ch[col*ldc + row]); f16 path only

__global__ void gemm64_kernel(const h16* __restrict__ A,
                              const h16* __restrict__ W,
                              h16* __restrict__ Ch, float* __restrict__ Cf,
                              const float* __restrict__ bias,
                              int M, int N, int K,
                              int lda, int ldc,
                              float alpha, int flags) {
  __shared__ __attribute__((aligned(16))) h16 Wlds[16384];  // up to 128x128 f16

  // ---- async-stage W (row-major, ld == K) into LDS
  const int wbytes = N * K * 2;
  for (int o = threadIdx.x * 16; o < wbytes; o += blockDim.x * 16)
    async_load_b128((unsigned int)(size_t)((char*)Wlds + o),
                    (unsigned long long)(size_t)((const char*)W + o));
  wait_async0();
  __syncthreads();

  const int tilesN = N >> 6;                       // 64-wide wave strips
  const int total  = (M >> 4) * tilesN;
  const int wave   = blockIdx.x * (blockDim.x >> 5) + (threadIdx.x >> 5);
  if (wave >= total) return;                       // wave-uniform

  const int tm = (wave / tilesN) << 4;
  const int tn = (wave % tilesN) << 6;

  v8f acc[4] = {{}, {}, {}, {}};
  for (int k = 0; k < K; k += 32) {
    v16h a = load_frag(A, tm, lda, k);
    v16h b[4];
#pragma unroll
    for (int t = 0; t < 4; ++t) b[t] = load_frag(Wlds, tn + (t << 4), K, k);
#pragma unroll
    for (int t = 0; t < 4; ++t) acc[t] = WMMA_F16(a, b[t], acc[t]);
  }

  const int lane = laneId();
  const int rb   = tm + ((lane >> 4) << 3);
#pragma unroll
  for (int t = 0; t < 4; ++t) {
    const int col = tn + (t << 4) + (lane & 15);
    const float bv = bias ? bias[col] : 0.0f;
#pragma unroll
    for (int v = 0; v < 8; ++v) {
      float y = acc[t][v] * alpha + bv;
      if (flags & GF_SIGMOID) y = 1.0f / (1.0f + __expf(-y));
      const int r = rb + v;
      if (flags & GF_TRANSPOSE) {
        Ch[(size_t)col * ldc + r] = (h16)y;
      } else {
        if (Ch) Ch[(size_t)r * ldc + col] = (h16)y;
        if (Cf) Cf[(size_t)r * ldc + col] = y;
      }
    }
  }
}

// ---------------------------------------------------------------------------
// 4b) Fused q/g GEMM: Q = qscale*(A Wq^T), G = sigmoid(A Wg^T + bg)
//     Shares the A (xn) stream: 8 WMMAs per A fragment, N = K = 128.
// ---------------------------------------------------------------------------
__global__ void gemm_qg_kernel(const h16* __restrict__ A,
                               const h16* __restrict__ Wq,
                               const h16* __restrict__ Wg,
                               h16* __restrict__ Q, h16* __restrict__ G,
                               const float* __restrict__ bg,
                               int M, float qscale) {
  __shared__ __attribute__((aligned(16))) h16 Wq_lds[16384];  // 128x128 f16
  __shared__ __attribute__((aligned(16))) h16 Wg_lds[16384];  // 128x128 f16

  for (int o = threadIdx.x * 16; o < 32768; o += blockDim.x * 16) {
    async_load_b128((unsigned int)(size_t)((char*)Wq_lds + o),
                    (unsigned long long)(size_t)((const char*)Wq + o));
    async_load_b128((unsigned int)(size_t)((char*)Wg_lds + o),
                    (unsigned long long)(size_t)((const char*)Wg + o));
  }
  wait_async0();
  __syncthreads();

  const int total = (M >> 4) * 2;                  // two 64-wide strips per row
  const int wave  = blockIdx.x * (blockDim.x >> 5) + (threadIdx.x >> 5);
  if (wave >= total) return;                       // wave-uniform

  const int tm = (wave >> 1) << 4;
  const int tn = (wave & 1) << 6;

  v8f accq[4] = {{}, {}, {}, {}};
  v8f accg[4] = {{}, {}, {}, {}};
  for (int k = 0; k < 128; k += 32) {
    v16h a = load_frag(A, tm, 128, k);
    v16h bq[4], bgf[4];
#pragma unroll
    for (int t = 0; t < 4; ++t) bq[t]  = load_frag(Wq_lds, tn + (t << 4), 128, k);
#pragma unroll
    for (int t = 0; t < 4; ++t) bgf[t] = load_frag(Wg_lds, tn + (t << 4), 128, k);
#pragma unroll
    for (int t = 0; t < 4; ++t) accq[t] = WMMA_F16(a, bq[t],  accq[t]);
#pragma unroll
    for (int t = 0; t < 4; ++t) accg[t] = WMMA_F16(a, bgf[t], accg[t]);
  }

  const int lane = laneId();
  const int rb   = tm + ((lane >> 4) << 3);
#pragma unroll
  for (int t = 0; t < 4; ++t) {
    const int col = tn + (t << 4) + (lane & 15);
    const float bgv = bg[col];
#pragma unroll
    for (int v = 0; v < 8; ++v) {
      const int r = rb + v;
      Q[(size_t)r * 128 + col] = (h16)(accq[t][v] * qscale);
      const float y = accg[t][v] + bgv;
      G[(size_t)r * 128 + col] = (h16)(1.0f / (1.0f + __expf(-y)));
    }
  }
}

// ---------------------------------------------------------------------------
// 5) tri[h][n][m] = sum_d xb[n][h*16+d] * zb[m][h*16+d]   (K=16, VALU)
// ---------------------------------------------------------------------------
__global__ void tri_kernel(const float* __restrict__ xb,
                           const float* __restrict__ zb,
                           float* __restrict__ tri) {
  const int n = blockIdx.x;   // 0..255
  const int h = blockIdx.y;   // 0..3
  const int m = threadIdx.x;  // 0..255
  const float* xr = xb + n * 64 + h * 16;
  const float* zr = zb + m * 64 + h * 16;
  float acc = 0.0f;
#pragma unroll
  for (int d = 0; d < 16; ++d) acc += xr[d] * zr[d];
  tri[((size_t)(h * 256 + n)) * 256 + m] = acc;
}

// ---------------------------------------------------------------------------
// 6) Fused attention core per (qtile, row); ALL 4 heads per block.
//    Phase 1: 8 waves x 8 logits tiles (wave's head fixed -> A frag reused)
//    Phase 2: softmax over M=256 for 64 (head,q) rows
//    Phase 3: 8 waves == 8 (head x c-tile) O tiles, P from LDS (ds_load)
// ---------------------------------------------------------------------------
__global__ void attn_kernel(const h16* __restrict__ qh,   // [65536][128] scaled
                            const h16* __restrict__ kh,   // [256][128]
                            const h16* __restrict__ vT,   // [128][256]
                            const h16* __restrict__ gh,   // [65536][128] sigmoid'd
                            const float* __restrict__ tri,// [4][256][256]
                            const float* __restrict__ mask,// [256][256]
                            h16* __restrict__ og) {       // [65536][128]
  __shared__ __attribute__((aligned(16))) float S[4][16][256];  // 64KB
  __shared__ __attribute__((aligned(16))) h16   P[4][16][256];  // 32KB
  __shared__ float red[64][4];
  __shared__ float rowmax[64];
  __shared__ float rowsum[64];

  const int qt  = blockIdx.x;   // 0..15  (16 q rows each)
  const int r   = blockIdx.y;   // 0..255
  const int tid = threadIdx.x;  // 256 threads = 8 waves
  const int wave = tid >> 5;
  const int lane = laneId();
  const int q0 = qt << 4;

  // ---- Phase 1: logits. wave -> head (wave>>1), 8 m-tiles each.
  {
    const int h = wave >> 1;
    const h16* qbase = qh + ((size_t)(r * 256 + q0)) * 128 + h * 32;
    const h16* kbase = kh + h * 32;
    const v16h afrag = load_frag(qbase, 0, 128, 0);   // K = CH = 32: one frag
#pragma unroll
    for (int i = 0; i < 8; ++i) {
      const int mt = ((wave & 1) << 3) + i;           // 0..15
      v16h b = load_frag(kbase, mt << 4, 128, 0);
      v8f acc = {};
      acc = WMMA_F16(afrag, b, acc);
      const int m  = (mt << 4) + (lane & 15);
      const int rb = (lane >> 4) << 3;
#pragma unroll
      for (int v = 0; v < 8; ++v) {
        const int qi = rb + v;
        const int qg = q0 + qi;
        const float bias = 1.0e9f * (mask[r * 256 + qg] - 1.0f)
                         + tri[((size_t)(h * 256 + qg)) * 256 + m];
        S[h][qi][m] = acc[v] + bias;
      }
    }
  }
  __syncthreads();

  // ---- Phase 2: softmax over m. 64 rows (h,qi), 4 threads per row.
  {
    const int R = tid >> 2;          // 0..63
    const int j = tid & 3;
    float* Srow = &S[R >> 4][R & 15][0];
    float mx = -3.0e38f;
#pragma unroll
    for (int i = 0; i < 64; ++i) mx = fmaxf(mx, Srow[j + (i << 2)]);
    red[R][j] = mx;
    __syncthreads();
    if (j == 0) {
      float m2 = fmaxf(fmaxf(red[R][0], red[R][1]), fmaxf(red[R][2], red[R][3]));
      rowmax[R] = m2;
    }
    __syncthreads();
    const float rm = rowmax[R];
    float sum = 0.0f;
#pragma unroll
    for (int i = 0; i < 64; ++i) {
      const int m = j + (i << 2);
      const float e = __expf(Srow[m] - rm);
      Srow[m] = e;
      sum += e;
    }
    red[R][j] = sum;
    __syncthreads();
    if (j == 0) rowsum[R] = red[R][0] + red[R][1] + red[R][2] + red[R][3];
    __syncthreads();
    const float inv = 1.0f / rowsum[R];
    h16* Prow = &P[R >> 4][R & 15][0];
#pragma unroll
    for (int i = 0; i < 64; ++i) {
      const int m = j + (i << 2);
      Prow[m] = (h16)(Srow[m] * inv);
    }
  }
  __syncthreads();

  // ---- Phase 3: O = P_h (16x256) @ V_h (256x32). wave -> (h, c-tile).
  {
    const int h  = wave >> 1;
    const int ct = wave & 1;
    const h16* vbase = vT + (size_t)(h * 32 + ct * 16) * 256;
    v8f acc = {};
#pragma unroll
    for (int k = 0; k < 256; k += 32) {
      v16h a = load_frag(&P[h][0][0], 0, 256, k);     // ds_load path
      v16h b = load_frag(vbase, 0, 256, k);
      acc = WMMA_F16(a, b, acc);
    }
    const int col = lane & 15;
    const int rb  = (lane >> 4) << 3;
#pragma unroll
    for (int v = 0; v < 8; ++v) {
      const int qi = rb + v;
      const size_t idx = ((size_t)(r * 256 + q0 + qi)) * 128
                       + h * 32 + ct * 16 + col;
      const float g = (float)gh[idx];
      og[idx] = (h16)(acc[v] * g);
    }
  }
}

// ---------------------------------------------------------------------------
// Host-side orchestration
// ---------------------------------------------------------------------------
extern "C" void kernel_launch(void* const* d_in, const int* in_sizes, int n_in,
                              void* d_out, int out_size, void* d_ws, size_t ws_size,
                              hipStream_t stream) {
  (void)in_sizes; (void)n_in; (void)out_size; (void)ws_size;

  const float* x    = (const float*)d_in[0];   // (1,256,256,128)
  const float* z    = (const float*)d_in[1];   // (1,256,128)
  const float* mask = (const float*)d_in[2];   // (1,256,256)
  const float* ln_w = (const float*)d_in[3];
  const float* ln_b = (const float*)d_in[4];
  const float* wx   = (const float*)d_in[5];   // (64,128)
  const float* wz   = (const float*)d_in[6];   // (64,128)
  const float* wq   = (const float*)d_in[7];   // (128,128)
  const float* wk   = (const float*)d_in[8];
  const float* wv   = (const float*)d_in[9];
  const float* wg   = (const float*)d_in[10];
  const float* bg   = (const float*)d_in[11];  // (128,)
  const float* wo   = (const float*)d_in[12];  // (128,128)
  const float* bo   = (const float*)d_in[13];  // (128,)
  float* out = (float*)d_out;                  // (1,256,256,128)

  // ---- workspace carve-up (256B aligned regions)
  char* ws = (char*)d_ws;
  size_t off = 0;
  auto carve = [&](size_t bytes) -> char* {
    char* p = ws + off;
    off = (off + bytes + 255) & ~(size_t)255;
    return p;
  };
  const size_t NROWS = 65536;                 // 256*256
  h16*   xn_h   = (h16*)  carve(NROWS * 128 * 2);
  h16*   qh     = (h16*)  carve(NROWS * 128 * 2);
  h16*   gh     = (h16*)  carve(NROWS * 128 * 2);
  h16*   og     = (h16*)  carve(NROWS * 128 * 2);
  float* tri    = (float*)carve(4 * 256 * 256 * 4);
  h16*   z_h    = (h16*)  carve(256 * 128 * 2);
  h16*   kh     = (h16*)  carve(256 * 128 * 2);
  h16*   vT     = (h16*)  carve(128 * 256 * 2);
  h16*   xbar_h = (h16*)  carve(256 * 128 * 2);
  h16*   wq_h   = (h16*)  carve(128 * 128 * 2);
  h16*   wk_h   = (h16*)  carve(128 * 128 * 2);
  h16*   wv_h   = (h16*)  carve(128 * 128 * 2);
  h16*   wg_h   = (h16*)  carve(128 * 128 * 2);
  h16*   wo_h   = (h16*)  carve(128 * 128 * 2);
  h16*   wx_h   = (h16*)  carve(64 * 128 * 2);
  h16*   wz_h   = (h16*)  carve(64 * 128 * 2);
  float* xb_f   = (float*)carve(256 * 64 * 4);
  float* zb_f   = (float*)carve(256 * 64 * 4);

  // ---- 1) layernorm (wave per row)
  layernorm_kernel<<<NROWS / 8, 256, 0, stream>>>(x, ln_w, ln_b, xn_h);

  // ---- 2) convert weights + z to f16
  auto cvt = [&](const float* src, h16* dst, int n) {
    cvt_f16_kernel<<<(n + 255) / 256, 256, 0, stream>>>(src, dst, n);
  };
  cvt(z,  z_h,  256 * 128);
  cvt(wq, wq_h, 128 * 128);
  cvt(wk, wk_h, 128 * 128);
  cvt(wv, wv_h, 128 * 128);
  cvt(wg, wg_h, 128 * 128);
  cvt(wo, wo_h, 128 * 128);
  cvt(wx, wx_h, 64 * 128);
  cvt(wz, wz_h, 64 * 128);

  // ---- 3) column mean of xn
  colmean_kernel<<<256, 128, 0, stream>>>(xn_h, xbar_h);

  // ---- 4) GEMMs: 16x64 strip per wave, W async-staged to LDS
  auto gemm = [&](const h16* A, const h16* W, h16* Ch, float* Cf,
                  const float* bias, int M, int N, int K,
                  int lda, int ldc, float alpha, int flags) {
    int strips = (M >> 4) * (N >> 6);
    int blocks = (strips + 7) / 8;
    gemm64_kernel<<<blocks, 256, 0, stream>>>(A, W, Ch, Cf, bias,
                                              M, N, K, lda, ldc, alpha, flags);
  };
  const float qscale = 0.17677669529663687f;  // CH^-0.5 = 1/sqrt(32)

  gemm(z_h, wk_h, kh, nullptr, nullptr, 256, 128, 128, 128, 128, 1.0f, 0);
  gemm(z_h, wv_h, vT, nullptr, nullptr, 256, 128, 128, 128, 256, 1.0f,
       GF_TRANSPOSE);                                         // vT[c][m]
  gemm(z_h,    wz_h, nullptr, zb_f, nullptr, 256, 64, 128, 128, 64, 1.0f, 0);
  gemm(xbar_h, wx_h, nullptr, xb_f, nullptr, 256, 64, 128, 128, 64, 1.0f, 0);

  // fused q/g projection: reads xn once, writes qh (scaled) + gh (sigmoid)
  {
    int strips = ((int)NROWS >> 4) * 2;
    gemm_qg_kernel<<<strips / 8, 256, 0, stream>>>(xn_h, wq_h, wg_h,
                                                   qh, gh, bg,
                                                   (int)NROWS, qscale);
  }

  // ---- 5) triangle bias
  tri_kernel<<<dim3(256, 4), 256, 0, stream>>>(xb_f, zb_f, tri);

  // ---- 6) fused attention (all heads per block)
  attn_kernel<<<dim3(16, 256), 256, 0, stream>>>(qh, kh, vT, gh, tri, mask, og);

  // ---- 7) output projection: out = og @ wo^T + bo  (f32 out)
  gemm(og, wo_h, nullptr, out, bo, NROWS, 128, 128, 128, 128, 1.0f, 0);
}
